// PrototypeBank_68324339745325
// MI455X (gfx1250) — matrix operand order repaired and checked by reading, not compiled, not run on hardware
//
#include <hip/hip_runtime.h>

typedef __attribute__((ext_vector_type(16))) _Float16 v16h;
typedef __attribute__((ext_vector_type(8)))  _Float16 v8h;
typedef __attribute__((ext_vector_type(8)))  float    v8f;

#define PB_EPS 1e-12f
#define PB_MOM 0.9f

// ---------------- block reduction (256 threads, wave32) ----------------
__device__ __forceinline__ float block_reduce_sum(float v, float* sbuf) {
    const int t = threadIdx.x;
    #pragma unroll
    for (int off = 16; off > 0; off >>= 1)
        v += __shfl_down(v, off, 32);
    if ((t & 31) == 0) sbuf[t >> 5] = v;
    __syncthreads();
    float r = 0.0f;
    if (t < 8) r = sbuf[t];
    if (t < 32) {
        #pragma unroll
        for (int off = 4; off > 0; off >>= 1)
            r += __shfl_down(r, off, 32);
        if (t == 0) sbuf[0] = r;
    }
    __syncthreads();
    float res = sbuf[0];
    __syncthreads();
    return res;
}

// ---------------- phase 2: normalize rows, accumulate segment sums ----------------
__global__ __launch_bounds__(256) void pb_norm_accum(
    const float* __restrict__ feats, const int* __restrict__ labels,
    _Float16* __restrict__ fh, float* __restrict__ sums,
    float* __restrict__ counts, int D) {
    __shared__ float sbuf[8];
    const int row = blockIdx.x;
    const int t = threadIdx.x;
    const float* fr = feats + (size_t)row * D;

    float ss = 0.0f;
    for (int d = 2 * t; d < D; d += 2 * 256) {
        float2 v = *(const float2*)(fr + d);
        ss += v.x * v.x + v.y * v.y;
    }
    float nrm = sqrtf(block_reduce_sum(ss, sbuf));
    float inv = 1.0f / fmaxf(nrm, PB_EPS);

    const int lab = labels[row];
    float* srow = sums + (size_t)lab * D;
    _Float16* frh = fh + (size_t)row * D;
    for (int d = 2 * t; d < D; d += 2 * 256) {
        float2 v = *(const float2*)(fr + d);
        float a = v.x * inv, b = v.y * inv;
        union { _Float16 h[2]; unsigned u; } p;
        p.h[0] = (_Float16)a; p.h[1] = (_Float16)b;
        *(unsigned*)(frh + d) = p.u;
        atomicAdd(srow + d + 0, a);
        atomicAdd(srow + d + 1, b);
    }
    if (t == 0) atomicAdd(counts + lab, 1.0f);
}

// ---------------- phase 3: EMA update + renormalize prototypes ----------------
__global__ __launch_bounds__(256) void pb_proto_update(
    const float* __restrict__ protos, const float* __restrict__ sums,
    const float* __restrict__ counts, _Float16* __restrict__ ph, int D) {
    __shared__ float sbuf[8];
    const int c = blockIdx.x;
    const int t = threadIdx.x;
    const float cnt = counts[c];
    const float* pr = protos + (size_t)c * D;
    const float* sr = sums + (size_t)c * D;
    const float mscale = (1.0f - PB_MOM) / fmaxf(cnt, 1.0f);

    float ss = 0.0f;
    for (int d = 2 * t; d < D; d += 2 * 256) {
        float2 p = *(const float2*)(pr + d);
        float2 s = *(const float2*)(sr + d);
        float a = PB_MOM * p.x + s.x * mscale;
        float b = PB_MOM * p.y + s.y * mscale;
        ss += a * a + b * b;
    }
    float nrm = sqrtf(block_reduce_sum(ss, sbuf));
    float inv = 1.0f / fmaxf(nrm, PB_EPS);
    const bool present = cnt > 0.0f;

    _Float16* prh = ph + (size_t)c * D;
    for (int d = 2 * t; d < D; d += 2 * 256) {
        float2 p = *(const float2*)(pr + d);
        float2 s = *(const float2*)(sr + d);
        float a = PB_MOM * p.x + s.x * mscale;
        float b = PB_MOM * p.y + s.y * mscale;
        a = present ? a * inv : p.x;
        b = present ? b * inv : p.y;
        union { _Float16 h[2]; unsigned u; } q;
        q.h[0] = (_Float16)a; q.h[1] = (_Float16)b;
        *(unsigned*)(prh + d) = q.u;
    }
}

// ---------------- phase 4: sim = f @ P'^T via v_wmma_f32_16x16x32_f16 ----------------
// grid.x = N/32 (32-row M blocks), grid.y = C/1024; 256 threads = 8 waves.
// Each wave: 2 M-tiles (shared across waves -> WGP$ hits) x 8 N-tiles (128 cols),
// K looped by 32. 16 WMMA per k-step for 10 tile loads.
__global__ __launch_bounds__(256) void pb_gemm(
    const _Float16* __restrict__ fh, const _Float16* __restrict__ ph,
    float* __restrict__ out, int C, int D) {
    const int lane = threadIdx.x & 31;
    const int wave = threadIdx.x >> 5;
    const int mbase = blockIdx.x * 32;
    const int cbase = blockIdx.y * 1024 + wave * 128;
    const int half = lane >> 4;     // 0: lanes 0-15, 1: lanes 16-31
    const int mn   = lane & 15;     // M row (for A) / N col (for B,D)

    const _Float16* arow0 = fh + (size_t)(mbase + mn) * D;
    const _Float16* arow1 = fh + (size_t)(mbase + 16 + mn) * D;

    v8f acc[2][8];
    const v8f vzero = {0.f, 0.f, 0.f, 0.f, 0.f, 0.f, 0.f, 0.f};
    #pragma unroll
    for (int mt = 0; mt < 2; ++mt)
        #pragma unroll
        for (int i = 0; i < 8; ++i) acc[mt][i] = vzero;

    for (int k0 = 0; k0 < D; k0 += 32) {
        // A tiles (16x32 f16): lane half h holds K=[k0+8h,+8) and [k0+16+8h,+8)
        v8h a0lo = *(const v8h*)(arow0 + k0 + half * 8);
        v8h a0hi = *(const v8h*)(arow0 + k0 + 16 + half * 8);
        v8h a1lo = *(const v8h*)(arow1 + k0 + half * 8);
        v8h a1hi = *(const v8h*)(arow1 + k0 + 16 + half * 8);
        v16h A0, A1;
        #pragma unroll
        for (int i = 0; i < 8; ++i) {
            A0[i] = a0lo[i]; A0[8 + i] = a0hi[i];
            A1[i] = a1lo[i]; A1[8 + i] = a1hi[i];
        }

        #pragma unroll
        for (int ct = 0; ct < 8; ++ct) {
            // B tile (32x16 f16): lane half h holds K=[k0+8h,+8) and [k0+16+8h,+8) of col mn
            const _Float16* brow =
                ph + (size_t)(cbase + ct * 16 + mn) * D + k0 + half * 8;
            v8h blo = *(const v8h*)(brow);
            v8h bhi = *(const v8h*)(brow + 16);
            v16h B;
            #pragma unroll
            for (int i = 0; i < 8; ++i) { B[i] = blo[i]; B[8 + i] = bhi[i]; }

            acc[0][ct] = __builtin_amdgcn_wmma_f32_16x16x32_f16(
                false, A0, false, B, (short)0, acc[0][ct], false, false);
            acc[1][ct] = __builtin_amdgcn_wmma_f32_16x16x32_f16(
                false, A1, false, B, (short)0, acc[1][ct], false, false);
        }
    }

    // D layout: VGPR r, lane -> M = mbase + mt*16 + r + 8*half, N = cbase + ct*16 + mn
    // Output is write-once/streaming (256 MB): non-temporal to protect L2/WGP$.
    #pragma unroll
    for (int mt = 0; mt < 2; ++mt) {
        #pragma unroll
        for (int ct = 0; ct < 8; ++ct) {
            const int col = cbase + ct * 16 + mn;
            #pragma unroll
            for (int r = 0; r < 8; ++r) {
                __builtin_nontemporal_store(
                    acc[mt][ct][r],
                    out + (size_t)(mbase + mt * 16 + r + 8 * half) * C + col);
            }
        }
    }
}

extern "C" void kernel_launch(void* const* d_in, const int* in_sizes, int n_in,
                              void* d_out, int out_size, void* d_ws, size_t ws_size,
                              hipStream_t stream) {
    (void)n_in; (void)out_size; (void)ws_size;
    const float* feats  = (const float*)d_in[0];
    const float* protos = (const float*)d_in[1];
    const int*   labels = (const int*)d_in[2];

    const int ND = in_sizes[0];      // N*D
    const int CD = in_sizes[1];      // C*D
    const int N  = in_sizes[2];
    const int D  = ND / N;           // 512
    const int C  = CD / D;           // 1024

    char* w = (char*)d_ws;
    _Float16* fh     = (_Float16*)w;                       // N*D f16
    _Float16* ph     = (_Float16*)(w + (size_t)ND * 2);    // C*D f16
    float*    sums   = (float*)(w + (size_t)ND * 2 + (size_t)CD * 2); // C*D f32
    float*    counts = sums + CD;                          // C f32

    hipMemsetAsync(sums, 0, (size_t)(CD + C) * sizeof(float), stream);

    pb_norm_accum<<<N, 256, 0, stream>>>(feats, labels, fh, sums, counts, D);
    pb_proto_update<<<C, 256, 0, stream>>>(protos, sums, counts, ph, D);

    dim3 grid(N / 32, C / 1024);
    pb_gemm<<<grid, 256, 0, stream>>>(fh, ph, (float*)d_out, C, D);
}
